// MambaBlock_52003464020733
// MI455X (gfx1250) — compile-verified, matrix operand order checked by reading, not run on hardware
//
#include <hip/hip_runtime.h>
#include <hip/hip_bf16.h>
#include <cstdint>

// ---------------- problem constants (from reference) ----------------
#define B_      4
#define T_      4096
#define C_      1024
#define DSTATE  16
#define DCONV   4
#define DINNER  2048      // EXPAND * C
#define DTRANK  64        // ceil(C/16)
#define NROWS   (B_ * T_) // 16384
#define XDBL_W  (DTRANK + 2 * DSTATE)   // 96

typedef __bf16 bf16_t;
typedef bf16_t v16bf __attribute__((ext_vector_type(16)));
typedef float  v8f   __attribute__((ext_vector_type(8)));

__device__ inline float sigmoidf_(float v) { return 1.f / (1.f + __expf(-v)); }

// ============================================================================
// WMMA GEMM: out = act(A * W^T + bias)
// A: M x K bf16 (row stride lda). W: N x K bf16 (torch Linear layout).
// Cf: optional f32 out (M x N, stride ldc). Cb: optional bf16 out.
// act: 0 none, 1 silu, 2 softplus, 3 leakyrelu(0.2)
// 256 thr = 8 waves (4M x 2N); block tile 128x128, K-step 32; per wave 32x64
// = 2x4 v_wmma_f32_16x16x32_bf16 (A frags reused 4x, B frags 2x).
// LDS tiles stored in fragment order:
//   subtile(16 rows)*512 + lane*16 + j, lane=(row&15)|(g<<4),
//   g=(k>>3)&1, j=((k>>4)<<3)|(k&7)  -> fragment = 1 contiguous v16bf / lane.
// Double-buffered LDS, register-prefetched global loads, branch-free guards.
// ============================================================================
#define TM 128
#define TN 128
#define TK 32

__global__ __launch_bounds__(256)
void gemm_bf16_wmma(const bf16_t* __restrict__ A, int lda,
                    const bf16_t* __restrict__ W,
                    const float* __restrict__ bias,
                    float* __restrict__ Cf, bf16_t* __restrict__ Cb, int ldc,
                    int M, int N, int K, int act)
{
  __shared__ alignas(64) bf16_t sA[2][(TM / 16) * 512];
  __shared__ alignas(64) bf16_t sB[2][(TN / 16) * 512];

  const int tid   = threadIdx.x;
  const int lane  = tid & 31;
  const int wave  = tid >> 5;
  const int waveM = wave >> 1;   // 0..3 -> rows waveM*32
  const int waveN = wave & 1;    // 0..1 -> cols waveN*64
  const int m_l   = lane & 15;
  const int grp   = lane >> 4;
  const int bm    = blockIdx.x * TM;
  const int bn    = blockIdx.y * TN;

  // ---- staging: 8 bf16 (16B) per chunk; A: 512 chunks, W: 512 chunks ----
  const int cA0 = tid * 2, cA1 = tid * 2 + 1;
  const int rA0 = cA0 >> 2, pA0 = cA0 & 3;
  const int rA1 = cA1 >> 2, pA1 = cA1 & 3;
  const int rW0 = rA0, pW0 = pA0;           // same mapping for 128-row W tile
  const int rW1 = rA1, pW1 = pA1;

  auto ldsOff = [](int r, int p) -> int {   // fragment-order LDS offset
    int g = p & 1, jb = (p >> 1) << 3;
    return (r >> 4) * 512 + (((r & 15) | (g << 4)) << 4) + jb;
  };
  const int oA0 = ldsOff(rA0, pA0);
  const int oA1 = ldsOff(rA1, pA1);
  const int oW0 = ldsOff(rW0, pW0);
  const int oW1 = ldsOff(rW1, pW1);
  // branch-free OOB handling: clamp row (address always legal), zero via select
  const bool w0Valid = (bn + rW0) < N;
  const bool w1Valid = (bn + rW1) < N;
  const int  rW0c = w0Valid ? (bn + rW0) : (N - 1);
  const int  rW1c = w1Valid ? (bn + rW1) : (N - 1);

  const int nk = K / TK;
  uint4 ra0, ra1, rw0, rw1;
  const uint4 z4 = make_uint4(0u, 0u, 0u, 0u);

  auto loadTiles = [&](int kt) {
    const int k0 = kt * TK;
    ra0 = *reinterpret_cast<const uint4*>(A + (size_t)(bm + rA0) * lda + k0 + pA0 * 8);
    ra1 = *reinterpret_cast<const uint4*>(A + (size_t)(bm + rA1) * lda + k0 + pA1 * 8);
    uint4 t0 = *reinterpret_cast<const uint4*>(W + (size_t)rW0c * K + k0 + pW0 * 8);
    uint4 t1 = *reinterpret_cast<const uint4*>(W + (size_t)rW1c * K + k0 + pW1 * 8);
    rw0 = w0Valid ? t0 : z4;
    rw1 = w1Valid ? t1 : z4;
  };
  auto storeTiles = [&](int buf) {
    *reinterpret_cast<uint4*>(&sA[buf][oA0]) = ra0;
    *reinterpret_cast<uint4*>(&sA[buf][oA1]) = ra1;
    *reinterpret_cast<uint4*>(&sB[buf][oW0]) = rw0;
    *reinterpret_cast<uint4*>(&sB[buf][oW1]) = rw1;
  };

  v8f acc[2][4];
  {
    v8f zero8 = {0.f, 0.f, 0.f, 0.f, 0.f, 0.f, 0.f, 0.f};
    #pragma unroll
    for (int tm = 0; tm < 2; ++tm)
      #pragma unroll
      for (int tn = 0; tn < 4; ++tn) acc[tm][tn] = zero8;
  }

  loadTiles(0);
  storeTiles(0);
  __syncthreads();

  for (int kt = 0; kt < nk; ++kt) {
    const int cur = kt & 1;
    const bool hasNext = (kt + 1) < nk;
    if (hasNext) loadTiles(kt + 1);          // prefetch while WMMA runs

    v16bf afrag[2], bfrag[4];
    #pragma unroll
    for (int tm = 0; tm < 2; ++tm)
      afrag[tm] = *reinterpret_cast<const v16bf*>(
          &sA[cur][(2 * waveM + tm) * 512 + lane * 16]);
    #pragma unroll
    for (int tn = 0; tn < 4; ++tn)
      bfrag[tn] = *reinterpret_cast<const v16bf*>(
          &sB[cur][(4 * waveN + tn) * 512 + lane * 16]);

    #pragma unroll
    for (int tm = 0; tm < 2; ++tm)
      #pragma unroll
      for (int tn = 0; tn < 4; ++tn)
        acc[tm][tn] = __builtin_amdgcn_wmma_f32_16x16x32_bf16(
            false, afrag[tm], false, bfrag[tn], (short)0, acc[tm][tn], false, false);

    if (hasNext) {
      storeTiles((kt + 1) & 1);              // other buffer: last read ended at
      __syncthreads();                       //   previous iteration's barrier
    }
  }

  // ---- epilogue: C/D layout row = grp*8 + r, col = m_l ----
  #pragma unroll
  for (int tm = 0; tm < 2; ++tm) {
    #pragma unroll
    for (int tn = 0; tn < 4; ++tn) {
      const int gn = bn + waveN * 64 + tn * 16 + m_l;
      if (gn < N) {
        const float bv = bias ? bias[gn] : 0.f;
        #pragma unroll
        for (int r = 0; r < 8; ++r) {
          const int gm = bm + waveM * 32 + tm * 16 + grp * 8 + r;
          float v = acc[tm][tn][r] + bv;
          if (act == 1)      v = v * sigmoidf_(v);
          else if (act == 2) v = (v > 20.f) ? v : log1pf(__expf(v));
          else if (act == 3) v = (v >= 0.f) ? v : 0.2f * v;
          const size_t idx = (size_t)gm * ldc + gn;
          if (Cf) Cf[idx] = v;
          if (Cb) Cb[idx] = (bf16_t)v;
        }
      }
    }
  }
}

// ---------------- f32 -> bf16 elementwise convert (weights, once) -----------
__global__ void cvt_bf16_kernel(const float* __restrict__ in, bf16_t* __restrict__ out, size_t n)
{
  size_t gid = (size_t)blockIdx.x * 256 + threadIdx.x;
  if (gid < n) out[gid] = (bf16_t)in[gid];
}

// ---------------- LayerNorm (one block per row), writes bf16 ----------------
__global__ __launch_bounds__(256)
void layernorm_kernel(const float* __restrict__ x, const float* __restrict__ g,
                      const float* __restrict__ b, bf16_t* __restrict__ y)
{
  __shared__ float red[256];
  const int row = blockIdx.x;
  const float* xr = x + (size_t)row * C_;
  float s = 0.f;
  for (int c = threadIdx.x; c < C_; c += 256) s += xr[c];
  red[threadIdx.x] = s; __syncthreads();
  for (int o = 128; o > 0; o >>= 1) { if (threadIdx.x < o) red[threadIdx.x] += red[threadIdx.x + o]; __syncthreads(); }
  float mu = red[0] * (1.f / C_);
  __syncthreads();
  float s2 = 0.f;
  for (int c = threadIdx.x; c < C_; c += 256) { float d = xr[c] - mu; s2 += d * d; }
  red[threadIdx.x] = s2; __syncthreads();
  for (int o = 128; o > 0; o >>= 1) { if (threadIdx.x < o) red[threadIdx.x] += red[threadIdx.x + o]; __syncthreads(); }
  float rstd = rsqrtf(red[0] * (1.f / C_) + 1e-5f);
  bf16_t* yr = y + (size_t)row * C_;
  for (int c = threadIdx.x; c < C_; c += 256)
    yr[c] = (bf16_t)((xr[c] - mu) * rstd * g[c] + b[c]);
}

// ---------------- sequence flip along T (bf16) ----------------
__global__ void flip_kernel(const bf16_t* __restrict__ in, bf16_t* __restrict__ out)
{
  size_t gid = (size_t)blockIdx.x * 256 + threadIdx.x;
  if (gid >= (size_t)NROWS * C_) return;
  int c = gid % C_;
  size_t n = gid / C_;
  int b = (int)(n / T_), t = (int)(n % T_);
  out[((size_t)b * T_ + (T_ - 1 - t)) * C_ + c] = in[gid];
}

// ---------- causal depthwise conv (width 4) + SiLU; f32 + bf16 out ----------
__global__ void conv_silu_kernel(const float* __restrict__ xz, const float* __restrict__ w,
                                 const float* __restrict__ bias,
                                 float* __restrict__ outf, bf16_t* __restrict__ outb)
{
  size_t gid = (size_t)blockIdx.x * 256 + threadIdx.x;
  if (gid >= (size_t)NROWS * DINNER) return;
  int d = gid % DINNER;
  size_t n = gid / DINNER;
  int b = (int)(n / T_), t = (int)(n % T_);
  float acc = bias[d];
  #pragma unroll
  for (int j = 0; j < DCONV; ++j) {
    int tt = t - (DCONV - 1) + j;
    if (tt >= 0) acc += w[d * DCONV + j] * xz[((size_t)b * T_ + tt) * (2 * DINNER) + d];
  }
  float v = acc * sigmoidf_(acc);
  outf[gid] = v;
  outb[gid] = (bf16_t)v;
}

// ---------- selective scan, fused D-skip + SiLU(z) gate; bf16 out -----------
__global__ __launch_bounds__(256)
void scan_kernel(const float* __restrict__ dt, const float* __restrict__ xdbl,
                 const float* __restrict__ xc, const float* __restrict__ xz,
                 const float* __restrict__ A_log, const float* __restrict__ Dp,
                 bf16_t* __restrict__ yg)
{
  const int d = blockIdx.x * 256 + threadIdx.x;   // channel
  const int b = blockIdx.y;                       // batch
  float As[DSTATE], h[DSTATE];
  #pragma unroll
  for (int s = 0; s < DSTATE; ++s) { As[s] = -__expf(A_log[d * DSTATE + s]); h[s] = 0.f; }
  const float Dd = Dp[d];

  for (int t = 0; t < T_; ++t) {
    size_t idx = (size_t)b * T_ + t;
    float dtv = dt[idx * DINNER + d];
    float xv  = xc[idx * DINNER + d];
    float zv  = xz[idx * (2 * DINNER) + DINNER + d];
    const float* row = xdbl + idx * XDBL_W;       // uniform -> scalar loads
    float dtx = dtv * xv;
    float y = 0.f;
    #pragma unroll
    for (int s = 0; s < DSTATE; ++s) {
      float dA = __expf(dtv * As[s]);
      h[s] = dA * h[s] + dtx * row[DTRANK + s];
      y += h[s] * row[DTRANK + DSTATE + s];
    }
    yg[idx * DINNER + d] = (bf16_t)((y + Dd * xv) * (zv * sigmoidf_(zv)));
  }
}

// -------- final combine: out = x_bm * (leaky(fc2) + x_fm) --------
__global__ void combine_kernel(const float* __restrict__ yb, const float* __restrict__ fc2o,
                               const float* __restrict__ yf, float* __restrict__ out)
{
  size_t gid = (size_t)blockIdx.x * 256 + threadIdx.x;
  if (gid >= (size_t)NROWS * C_) return;
  out[gid] = yb[gid] * (fc2o[gid] + yf[gid]);
}

// ---------------- launcher ----------------
static inline void launch_gemm(const bf16_t* A, int lda, const bf16_t* W, const float* bias,
                               float* Cf, bf16_t* Cb, int ldc, int M, int N, int K, int act,
                               hipStream_t stream)
{
  dim3 grid((unsigned)(M / TM), (unsigned)((N + TN - 1) / TN));
  gemm_bf16_wmma<<<grid, 256, 0, stream>>>(A, lda, W, bias, Cf, Cb, ldc, M, N, K, act);
}

static inline void launch_cvt(const float* in, bf16_t* out, size_t n, hipStream_t stream)
{
  cvt_bf16_kernel<<<(unsigned)((n + 255) / 256), 256, 0, stream>>>(in, out, n);
}

extern "C" void kernel_launch(void* const* d_in, const int* in_sizes, int n_in,
                              void* d_out, int out_size, void* d_ws, size_t ws_size,
                              hipStream_t stream)
{
  (void)in_sizes; (void)n_in; (void)out_size; (void)ws_size;
  const float* x     = (const float*)d_in[0];
  const float* ln_g  = (const float*)d_in[1];
  const float* ln_b  = (const float*)d_in[2];
  const float* fc1_w = (const float*)d_in[3];
  const float* fc1_b = (const float*)d_in[4];
  const float* fc2_w = (const float*)d_in[5];
  const float* fc2_b = (const float*)d_in[6];
  // mamba params at base 7 (fwd) / 16 (bwd):
  // in_proj_w, conv_w, conv_b, x_proj_w, dt_proj_w, dt_proj_b, A_log, D, out_proj_w

  // ---------- workspace carve-up ----------
  float* ws = (float*)d_ws;
  size_t o = 0;
  float* xzbuf = ws + o; o += (size_t)NROWS * 2 * DINNER;   // in_proj out (f32)
  float* xcbuf = ws + o; o += (size_t)NROWS * DINNER;       // conv out (f32)
  float* xdbl  = ws + o; o += (size_t)NROWS * XDBL_W;       // x_proj out (f32)
  float* dtbuf = ws + o; o += (size_t)NROWS * DINNER;       // dt (f32)
  float* y_f   = ws + o; o += (size_t)NROWS * C_;
  float* y_b   = ws + o; o += (size_t)NROWS * C_;
  float* fc2o  = ws + o; o += (size_t)NROWS * C_;

  bf16_t* bws = (bf16_t*)(ws + o);
  size_t ob = 0;
  bf16_t* x_ln_bf  = bws + ob; ob += (size_t)NROWS * C_;
  bf16_t* x_fc1_bf = bws + ob; ob += (size_t)NROWS * C_;
  bf16_t* x_rev_bf = bws + ob; ob += (size_t)NROWS * C_;
  bf16_t* xc_bf    = bws + ob; ob += (size_t)NROWS * DINNER;
  bf16_t* xdbl_bf  = bws + ob; ob += (size_t)NROWS * XDBL_W;
  bf16_t* yg_bf    = bws + ob; ob += (size_t)NROWS * DINNER;
  // bf16 weights
  bf16_t* wfc1 = bws + ob; ob += (size_t)C_ * C_;
  bf16_t* wfc2 = bws + ob; ob += (size_t)C_ * C_;
  bf16_t* winp[2]; bf16_t* wxp[2]; bf16_t* wdtp[2]; bf16_t* wout[2];
  for (int dir = 0; dir < 2; ++dir) {
    winp[dir] = bws + ob; ob += (size_t)(2 * DINNER) * C_;
    wxp[dir]  = bws + ob; ob += (size_t)XDBL_W * DINNER;
    wdtp[dir] = bws + ob; ob += (size_t)DINNER * DTRANK;
    wout[dir] = bws + ob; ob += (size_t)C_ * DINNER;
  }

  const size_t nElemsC  = (size_t)NROWS * C_;
  const size_t nElemsDI = (size_t)NROWS * DINNER;
  const unsigned blkC  = (unsigned)((nElemsC  + 255) / 256);
  const unsigned blkDI = (unsigned)((nElemsDI + 255) / 256);

  // ---------- weight conversion (cheap, once per launch) ----------
  launch_cvt(fc1_w, wfc1, (size_t)C_ * C_, stream);
  launch_cvt(fc2_w, wfc2, (size_t)C_ * C_, stream);
  for (int dir = 0; dir < 2; ++dir) {
    const int p = 7 + dir * 9;
    launch_cvt((const float*)d_in[p + 0], winp[dir], (size_t)(2 * DINNER) * C_, stream);
    launch_cvt((const float*)d_in[p + 3], wxp[dir],  (size_t)XDBL_W * DINNER,   stream);
    launch_cvt((const float*)d_in[p + 4], wdtp[dir], (size_t)DINNER * DTRANK,   stream);
    launch_cvt((const float*)d_in[p + 8], wout[dir], (size_t)C_ * DINNER,       stream);
  }

  // 1) LayerNorm -> bf16
  layernorm_kernel<<<NROWS, 256, 0, stream>>>(x, ln_g, ln_b, x_ln_bf);
  // 2) fc1 -> bf16 only (all consumers read bf16)
  launch_gemm(x_ln_bf, C_, wfc1, fc1_b, nullptr, x_fc1_bf, C_, NROWS, C_, C_, 0, stream);
  // 3) flip for backward path
  flip_kernel<<<blkC, 256, 0, stream>>>(x_fc1_bf, x_rev_bf);

  // 4) two mamba passes, shared scratch
  for (int dir = 0; dir < 2; ++dir) {
    const bf16_t* in   = dir ? x_rev_bf : x_fc1_bf;
    float*        yout = dir ? y_b : y_f;
    const int     p    = 7 + dir * 9;
    const float* conv_w    = (const float*)d_in[p + 1];
    const float* conv_b    = (const float*)d_in[p + 2];
    const float* dt_proj_b = (const float*)d_in[p + 5];
    const float* A_log     = (const float*)d_in[p + 6];
    const float* Dp        = (const float*)d_in[p + 7];

    launch_gemm(in, C_, winp[dir], nullptr, xzbuf, nullptr, 2 * DINNER,
                NROWS, 2 * DINNER, C_, 0, stream);
    conv_silu_kernel<<<blkDI, 256, 0, stream>>>(xzbuf, conv_w, conv_b, xcbuf, xc_bf);
    launch_gemm(xc_bf, DINNER, wxp[dir], nullptr, xdbl, xdbl_bf, XDBL_W,
                NROWS, XDBL_W, DINNER, 0, stream);
    launch_gemm(xdbl_bf, XDBL_W, wdtp[dir], dt_proj_b, dtbuf, nullptr, DINNER,
                NROWS, DINNER, DTRANK, 2 /*softplus*/, stream);
    scan_kernel<<<dim3(DINNER / 256, B_), 256, 0, stream>>>(
        dtbuf, xdbl, xcbuf, xzbuf, A_log, Dp, yg_bf);
    launch_gemm(yg_bf, DINNER, wout[dir], nullptr, yout, nullptr, C_,
                NROWS, C_, DINNER, 0, stream);
  }

  // 5) fc2(x_rev) with fused LeakyReLU(0.2)
  launch_gemm(x_rev_bf, C_, wfc2, fc2_b, fc2o, nullptr, C_, NROWS, C_, C_, 3, stream);
  // 6) out = x_bm * x_relu + x_fm * x_bm
  combine_kernel<<<blkC, 256, 0, stream>>>(y_b, fc2o, y_f, (float*)d_out);
}
